// GATrAdapter_2353642078924
// MI455X (gfx1250) — compile-verified, matrix operand order checked by reading, not run on hardware
//
#include <hip/hip_runtime.h>
#include <hip/hip_bf16.h>
#include <math.h>

typedef __attribute__((ext_vector_type(16))) _Float16 v16h;
typedef __attribute__((ext_vector_type(8)))  _Float16 v8h;
typedef __attribute__((ext_vector_type(8)))  float    v8f;
typedef __attribute__((ext_vector_type(4)))  float    v4f;

#define BT    4096      // total tokens (B * S * N)
#define TPB   2048      // tokens per batch
#define FDIM  160       // padded feature dim (128 mv + 24 s + 8 pad)
#define M1OUT 288       // padded m1 output (256 mv + 24 s + 8 pad)

// blade index -> bitmask (bit i = e_i present), canonical _BLADES order
__constant__ int BMASK[16] = {0,1,2,4,8,3,5,9,6,10,12,7,11,13,14,15};
// bitmask -> blade index
__constant__ int M2I[16]   = {0,1,2,5,3,6,8,11,4,7,9,12,10,13,14,15};

// ---------------- workspace layout (float offsets) ----------------
constexpr size_t W160     = 160 * 160;
constexpr size_t OFF_WQ   = 0;
constexpr size_t OFF_WK   = OFF_WQ  + W160;
constexpr size_t OFF_WV   = OFF_WK  + W160;
constexpr size_t OFF_WAO  = OFF_WV  + W160;
constexpr size_t OFF_WM1  = OFF_WAO + W160;
constexpr size_t OFF_WM2  = OFF_WM1 + (size_t)160 * M1OUT;
constexpr size_t OFF_WOUT = OFF_WM2 + W160;
constexpr size_t OFF_BAO  = OFF_WOUT + 160 * 16;
constexpr size_t OFF_BM1  = OFF_BAO + 160;
constexpr size_t OFF_BM2  = OFF_BM1 + M1OUT;
constexpr size_t OFF_BOUT = OFF_BM2 + 160;
constexpr size_t OFF_H    = ((OFF_BOUT + 16 + 31) / 32) * 32;
constexpr size_t OFF_N    = OFF_H + (size_t)BT * FDIM;
constexpr size_t OFF_Q    = OFF_N + (size_t)BT * FDIM;
constexpr size_t OFF_K    = OFF_Q + (size_t)BT * FDIM;
constexpr size_t OFF_V    = OFF_K + (size_t)BT * FDIM;
constexpr size_t OFF_O    = OFF_V + (size_t)BT * FDIM;
constexpr size_t OFF_G    = OFF_O + (size_t)BT * FDIM;
constexpr size_t OFF_U    = OFF_G + (size_t)BT * FDIM;
constexpr size_t OFF_END  = OFF_U + (size_t)BT * M1OUT;   // multiple of 32 floats

// f16 region (offsets in halves, based at ws + OFF_END)
constexpr size_t ATT_SZ = (size_t)2 * 8 * TPB * 32;  // q/k/vT buffers
constexpr size_t QAo  = 0;
constexpr size_t KAo  = QAo + ATT_SZ;
constexpr size_t VTo  = KAo + ATT_SZ;                // V transposed [b,h,dim,t]
constexpr size_t SW160 = (size_t)160 * 160;          // swizzled weight (halves)
constexpr size_t SW288 = (size_t)160 * M1OUT;
constexpr size_t WHQo  = VTo + ATT_SZ;
constexpr size_t WHKo  = WHQo + SW160;
constexpr size_t WHVo  = WHKo + SW160;
constexpr size_t WHAOo = WHVo + SW160;
constexpr size_t WHM1o = WHAOo + SW160;
constexpr size_t WHM2o = WHM1o + SW288;

struct Ptrs {
  const float *x, *w_in_mv, *b_in_mv, *w_in_m2s, *b_in_s,
      *w_qkv_mv, *w_qkv_s2mv, *w_qkv_m2s, *w_qkv_s2s,
      *w_ao_mv, *w_ao_s2mv, *w_ao_m2s, *w_ao_s2s, *b_ao_mv, *b_ao_s,
      *w_m1_mv, *w_m1_s2mv, *w_m1_m2s, *w_m1_s2s, *b_m1_mv, *b_m1_s,
      *w_m2_mv, *w_m2_s2mv, *w_m2_m2s, *w_m2_s2s, *b_m2_mv, *b_m2_s,
      *w_out_mv, *w_out_s2mv, *b_out_mv;
};

__device__ __forceinline__ int reorder_par(int a, int b) {
  int s = 0;
#pragma unroll
  for (int j = 0; j < 4; ++j)
    if ((b >> j) & 1) s += __popc(a >> (j + 1));
  return s & 1;
}
__device__ __forceinline__ float geluf(float x) {
  return 0.5f * x * (1.0f + erff(x * 0.70710678118654752f));
}
// K index held by (fragment slot idx, lane half) for 16-bit A/B fragments.
// Slots 0..7  -> k = half*8 + idx        (contiguous run 1)
// Slots 8..15 -> k = 16 + half*8 + idx-8 (contiguous run 2)
__device__ __forceinline__ int kmap(int idx, int half) {
  return (idx < 8 ? idx : 8 + idx) + half * 8;
}
__device__ __forceinline__ v8f wmma_f16(v16h a, v16h b, v8f c) {
  return __builtin_amdgcn_wmma_f32_16x16x32_f16(false, a, false, b, (short)0, c,
                                                false, false);
}
__device__ __forceinline__ v16h cat8(v8h lo, v8h hi) {
  return __builtin_shufflevector(lo, hi, 0, 1, 2, 3, 4, 5, 6, 7,
                                 8, 9, 10, 11, 12, 13, 14, 15);
}
// build A fragment from two contiguous float8 runs, converted to f16
__device__ __forceinline__ v16h frag_from_f32(const float* p0, const float* p1) {
  v4f a0 = *(const v4f*)(p0);
  v4f a1 = *(const v4f*)(p0 + 4);
  v4f b0 = *(const v4f*)(p1);
  v4f b1 = *(const v4f*)(p1 + 4);
  v16h r;
#pragma unroll
  for (int i = 0; i < 4; ++i) {
    r[i]      = (_Float16)a0[i];
    r[i + 4]  = (_Float16)a1[i];
    r[i + 8]  = (_Float16)b0[i];
    r[i + 12] = (_Float16)b1[i];
  }
  return r;
}

// ---------------- 1) build combined equi_linear matrices -----------------
__global__ void ga_prep(Ptrs P, float* ws, _Float16* hb) {
  int mid = blockIdx.x;
  const float *wmv = nullptr, *s2mv = nullptr, *m2s = nullptr, *s2s = nullptr,
              *bmv = nullptr, *bs = nullptr;
  float *W = nullptr, *Bd = nullptr;
  _Float16* Wh = nullptr;
  int O = 8, OUTP = 160, HSout = 24;
  switch (mid) {
    case 0: case 1: case 2: {
      int m = mid;
      wmv  = P.w_qkv_mv  + m * 576;  s2mv = P.w_qkv_s2mv + m * 192;
      m2s  = P.w_qkv_m2s + m * 192;  s2s  = P.w_qkv_s2s  + m * 576;
      W = ws + OFF_WQ + (size_t)m * W160;
      Wh = hb + WHQo + (size_t)m * SW160;
      break; }
    case 3:
      wmv = P.w_ao_mv; s2mv = P.w_ao_s2mv; m2s = P.w_ao_m2s; s2s = P.w_ao_s2s;
      bmv = P.b_ao_mv; bs = P.b_ao_s; W = ws + OFF_WAO; Bd = ws + OFF_BAO;
      Wh = hb + WHAOo;
      break;
    case 4:
      O = 16; OUTP = M1OUT;
      wmv = P.w_m1_mv; s2mv = P.w_m1_s2mv; m2s = P.w_m1_m2s; s2s = P.w_m1_s2s;
      bmv = P.b_m1_mv; bs = P.b_m1_s; W = ws + OFF_WM1; Bd = ws + OFF_BM1;
      Wh = hb + WHM1o;
      break;
    case 5:
      wmv = P.w_m2_mv; s2mv = P.w_m2_s2mv; m2s = P.w_m2_m2s; s2s = P.w_m2_s2s;
      bmv = P.b_m2_mv; bs = P.b_m2_s; W = ws + OFF_WM2; Bd = ws + OFF_BM2;
      Wh = hb + WHM2o;
      break;
    default:
      O = 1; OUTP = 16; HSout = 0;
      wmv = P.w_out_mv; s2mv = P.w_out_s2mv; bmv = P.b_out_mv;
      W = ws + OFF_WOUT; Bd = ws + OFF_BOUT;
      break;
  }
  int MVO = O * 16;
  int total = 160 * OUTP;
  for (int e = threadIdx.x; e < total; e += blockDim.x) {
    int f = e / OUTP, g = e % OUTP;
    float v = 0.f;
    if (f < 128 && g < MVO) {
      int i = f >> 4, q = f & 15, o = g >> 4, pc = g & 15;
      int mp = BMASK[pc], mq = BMASK[q];
      if (pc == q) v += wmv[(o * 8 + i) * 9 + __popc(mp)];             // grade proj
      if ((mp & 1) && mq == (mp & ~1))
        v += wmv[(o * 8 + i) * 9 + 5 + __popc(mq)];                    // e0 ^ blade
    } else if (f >= 128 && f < 152 && g < MVO) {
      if (((g & 15) == 0) && s2mv) v = s2mv[(g >> 4) * 24 + (f - 128)];
    } else if (f < 128 && g >= MVO && g < MVO + HSout) {
      if (((f & 15) == 0) && m2s) v = m2s[(g - MVO) * 8 + (f >> 4)];
    } else if (f >= 128 && f < 152 && g >= MVO && g < MVO + HSout) {
      if (s2s) v = s2s[(g - MVO) * 24 + (f - 128)];
    }
    W[(size_t)f * OUTP + g] = v;
  }
  if (Bd) {
    for (int g = threadIdx.x; g < OUTP; g += blockDim.x) {
      float bv = 0.f;
      if (g < MVO) { if (((g & 15) == 0) && bmv) bv = bmv[g >> 4]; }
      else if (g < MVO + HSout && bs) bv = bs[g - MVO];
      Bd[g] = bv;
    }
  }
  __syncthreads();
  if (Wh) {
    // swizzle to fragment order: Wh[((tn*5+kc)*32+lane)*16+idx]
    for (int e = threadIdx.x; e < total; e += blockDim.x) {
      int idx = e & 15;
      int lane = (e >> 4) & 31;
      int kc = (e >> 9) % 5;
      int tn = (e >> 9) / 5;
      int kk = kc * 32 + kmap(idx, lane >> 4);
      Wh[e] = (_Float16)W[(size_t)kk * OUTP + tn * 16 + (lane & 15)];
    }
  }
}

// ---------------- 2) embedding: x -> h (w_in equi_linear) ----------------
__global__ void ga_embed(Ptrs P, float* ws) {
  int tok = blockIdx.x * blockDim.x + threadIdx.x;
  if (tok >= BT) return;
  const float* xi = P.x + (size_t)tok * 6;
  float mv[16];
#pragma unroll
  for (int i = 0; i < 16; ++i) mv[i] = 0.f;
  mv[0] = 1.f;  mv[14] = 1.f;
  mv[13] = -xi[0]; mv[12] = xi[1]; mv[11] = -xi[2];
  mv[5] = 0.5f * xi[3]; mv[6] = 0.5f * xi[4]; mv[7] = 0.5f * xi[5];
  float* hr = ws + OFF_H + (size_t)tok * FDIM;
  for (int o = 0; o < 8; ++o) {
#pragma unroll
    for (int pc = 0; pc < 16; ++pc) {
      int mp = BMASK[pc];
      float v = P.w_in_mv[o * 9 + __popc(mp)] * mv[pc];
      if (mp & 1) {
        int mb = mp & ~1;
        v += P.w_in_mv[o * 9 + 5 + __popc(mb)] * mv[M2I[mb]];
      }
      if (pc == 0) v += P.b_in_mv[o];
      hr[o * 16 + pc] = v;
    }
  }
  for (int s = 0; s < 24; ++s) hr[128 + s] = P.w_in_m2s[s] + P.b_in_s[s];
  for (int f = 152; f < 160; ++f) hr[f] = 0.f;
}

// ---------------- 3) equi_norm -------------------------------------------
__global__ void ga_norm(const float* __restrict__ src, float* __restrict__ dst) {
  int tok = blockIdx.x * blockDim.x + threadIdx.x;
  if (tok >= BT) return;
  const float* s = src + (size_t)tok * FDIM;
  float* d = dst + (size_t)tok * FDIM;
  float sq = 0.f;
  for (int c = 0; c < 8; ++c)
#pragma unroll
    for (int q = 0; q < 16; ++q) {
      if (BMASK[q] & 1) continue;  // INNER mask: blades without e0
      float v = s[c * 16 + q];
      sq += v * v;
    }
  float im = 1.f / sqrtf(sq * (1.f / 8.f) + 0.01f);
  float ss = 0.f;
  for (int i = 0; i < 24; ++i) { float v = s[128 + i]; ss += v * v; }
  float is = 1.f / sqrtf(ss * (1.f / 24.f) + 0.01f);
  for (int f = 0; f < 128; ++f) d[f] = s[f] * im;
  for (int i = 0; i < 24; ++i) d[128 + i] = s[128 + i] * is;
  for (int f = 152; f < 160; ++f) d[f] = 0.f;
}

// ---------------- 4) WMMA GEMM: C[4096 x N] = A[4096 x 160] * W[160 x N]
template <int N, bool BIAS, bool RES>
__global__ void ga_gemm(const float* __restrict__ A, const _Float16* __restrict__ Wh,
                        const float* __restrict__ bias, const float* __restrict__ res,
                        float* __restrict__ C) {
  int wave = threadIdx.x >> 5, lane = threadIdx.x & 31;
  int tN = blockIdx.y * 4 + wave;
  if (tN * 16 >= N) return;
  int m = lane & 15, half = lane >> 4;
  int mrow = blockIdx.x * 16 + m;
  int ncol = tN * 16 + m;
  const float* Arow = A + (size_t)mrow * FDIM;
  const _Float16* Wb = Wh + ((size_t)tN * 5 * 32 + lane) * 16;
  v8f acc = {};
#pragma unroll
  for (int kc = 0; kc < 5; ++kc) {
    int k0 = kc * 32;
    v16h a = frag_from_f32(Arow + k0 + half * 8, Arow + k0 + 16 + half * 8);
    const _Float16* wp = Wb + (size_t)kc * 32 * 16;
    v16h b = cat8(*(const v8h*)(wp), *(const v8h*)(wp + 8));
    acc = wmma_f16(a, b, acc);
  }
  float bv = BIAS ? bias[ncol] : 0.f;
#pragma unroll
  for (int r = 0; r < 8; ++r) {
    int row = blockIdx.x * 16 + r + 8 * half;
    float v = acc[r] + bv;
    if (RES) v += res[(size_t)row * FDIM + ncol];  // only used when N==FDIM
    C[(size_t)row * N + ncol] = v;
  }
}

// ---------------- 5) repack q/k -> [b,h,t,32] f16 ; v -> [b,h,dim,t] f16 --
// V dim 31 is seeded with 1.0 so the PV matmul accumulates the softmax
// denominator for free in output column 15 of the second PV tile.
__global__ void ga_repack(float* ws, _Float16* hb) {
  int tok = blockIdx.x * blockDim.x + threadIdx.x;
  if (tok >= BT) return;
  int b = tok >> 11, t = tok & (TPB - 1);
  const float* qr = ws + OFF_Q + (size_t)tok * FDIM;
  const float* kr = ws + OFF_K + (size_t)tok * FDIM;
  const float* vr = ws + OFF_V + (size_t)tok * FDIM;
  float* orow = ws + OFF_O + (size_t)tok * FDIM;
  for (int f = 152; f < 160; ++f) orow[f] = 0.f;     // pad of attention output
  const float scale = 0.30151134457776363f;          // 1/sqrt(8*1 + 3)
  for (int hd = 0; hd < 8; ++hd) {
    size_t rowb = ((size_t)(b * 8 + hd) * TPB + t) * 32;
    _Float16* qa = hb + QAo + rowb;
    _Float16* ka = hb + KAo + rowb;
    _Float16* vt = hb + VTo + (size_t)(b * 8 + hd) * 32 * TPB;
#pragma unroll
    for (int d = 0; d < 32; ++d) {
      float qv = 0.f, kv = 0.f, vv = 0.f;
      if (d < 16) {
        float inner = (BMASK[d] & 1) ? 0.f : 1.f;    // fold INNER into Q
        qv = qr[hd * 16 + d] * inner * scale;
        kv = kr[hd * 16 + d];
        vv = vr[hd * 16 + d];
      } else if (d < 19) {
        int si = 128 + hd * 3 + (d - 16);
        qv = qr[si] * scale; kv = kr[si]; vv = vr[si];
      } else if (d == 31) {
        vv = 1.0f;                                   // ones column -> l
      }
      qa[d] = (_Float16)qv; ka[d] = (_Float16)kv;
      vt[(size_t)d * TPB + t] = (_Float16)vv;
    }
  }
}

// ---------------- 6) flash attention: 1 wave per (b, h, 16-query tile) ----
// 32 keys per main iteration (2 score WMMAs + 2 full-K PV WMMAs); the
// softmax denominator comes from the PV matmul (ones column in V).
__global__ void ga_attn(const _Float16* __restrict__ hb, float* __restrict__ ws) {
  int qb = blockIdx.x, hd = blockIdx.y, b = blockIdx.z;
  int lane = threadIdx.x;
  int m = lane & 15, half = lane >> 4;
  size_t hbase = (size_t)(b * 8 + hd) * TPB * 32;
  const _Float16* qa = hb + QAo + hbase;
  const _Float16* ka = hb + KAo + hbase;
  const _Float16* vt = hb + VTo + hbase;   // [dim][t]

  const _Float16* qrow = qa + (size_t)(qb * 16 + m) * 32;
  v16h qf = cat8(*(const v8h*)(qrow + half * 8),
                 *(const v8h*)(qrow + 16 + half * 8));
  const _Float16* v0row = vt + (size_t)m * TPB;         // dim = m
  const _Float16* v1row = vt + (size_t)(16 + m) * TPB;  // dim = 16+m

  v8f o0 = {}, o1 = {};
  float mr[8];
#pragma unroll
  for (int r = 0; r < 8; ++r) mr[r] = -1e30f;
  __shared__ alignas(16) _Float16 pb[16 * 32];   // P tile, 16 x 32
  v8f zacc = {};

  int nb = qb + 1;          // visible 16-key tiles (block-aligned causal mask)
  int nfull = nb >> 1;      // 32-key iterations
  for (int jj = 0; jj < nfull; ++jj) {
    int kb = jj * 32;
    const _Float16* krow0 = ka + (size_t)(kb + m) * 32;
    const _Float16* krow1 = ka + (size_t)(kb + 16 + m) * 32;
    v16h kf0 = cat8(*(const v8h*)(krow0 + half * 8),
                    *(const v8h*)(krow0 + 16 + half * 8));
    v16h kf1 = cat8(*(const v8h*)(krow1 + half * 8),
                    *(const v8h*)(krow1 + 16 + half * 8));
    if (jj + 1 < nfull) {   // prefetch next key tile (global_prefetch_b8)
      __builtin_prefetch((const void*)(ka + (size_t)(kb + 32 + m) * 32), 0, 1);
      __builtin_prefetch((const void*)(v0row + kb + 32), 0, 1);
    }
    v8f s0 = wmma_f16(qf, kf0, zacc);
    v8f s1 = wmma_f16(qf, kf1, zacc);
#pragma unroll
    for (int r = 0; r < 8; ++r) {
      float rowmax = fmaxf(s0[r], s1[r]);
      rowmax = fmaxf(rowmax, __shfl_xor(rowmax, 1));
      rowmax = fmaxf(rowmax, __shfl_xor(rowmax, 2));
      rowmax = fmaxf(rowmax, __shfl_xor(rowmax, 4));
      rowmax = fmaxf(rowmax, __shfl_xor(rowmax, 8));
      float mnew = fmaxf(mr[r], rowmax);
      float corr = __expf(mr[r] - mnew);
      mr[r] = mnew;
      o0[r] *= corr; o1[r] *= corr;
      int row = (r + 8 * half) * 32 + m;
      pb[row]      = (_Float16)__expf(s0[r] - mnew);
      pb[row + 16] = (_Float16)__expf(s1[r] - mnew);
    }
    __syncthreads();
    v16h pf  = cat8(*(const v8h*)(pb + m * 32 + half * 8),
                    *(const v8h*)(pb + m * 32 + 16 + half * 8));
    v16h vf0 = cat8(*(const v8h*)(v0row + kb + half * 8),
                    *(const v8h*)(v0row + kb + 16 + half * 8));
    v16h vf1 = cat8(*(const v8h*)(v1row + kb + half * 8),
                    *(const v8h*)(v1row + kb + 16 + half * 8));
    __syncthreads();
    o0 = wmma_f16(pf, vf0, o0);   // dims 0..15: mv comps
    o1 = wmma_f16(pf, vf1, o1);   // dims 16..18: scalars, dim31: l
  }
  if (nb & 1) {                    // leftover 16-key tile
    int kb = (nb - 1) * 16;
    const _Float16* krow = ka + (size_t)(kb + m) * 32;
    v16h kf = cat8(*(const v8h*)(krow + half * 8),
                   *(const v8h*)(krow + 16 + half * 8));
    v8f s = wmma_f16(qf, kf, zacc);
#pragma unroll
    for (int r = 0; r < 8; ++r) {
      float rowmax = s[r];
      rowmax = fmaxf(rowmax, __shfl_xor(rowmax, 1));
      rowmax = fmaxf(rowmax, __shfl_xor(rowmax, 2));
      rowmax = fmaxf(rowmax, __shfl_xor(rowmax, 4));
      rowmax = fmaxf(rowmax, __shfl_xor(rowmax, 8));
      float mnew = fmaxf(mr[r], rowmax);
      float corr = __expf(mr[r] - mnew);
      mr[r] = mnew;
      o0[r] *= corr; o1[r] *= corr;
      int row = (r + 8 * half) * 32 + m;
      pb[row]      = (_Float16)__expf(s[r] - mnew);
      pb[row + 16] = (_Float16)0.f;   // masked half: P = 0 nullifies V garbage
    }
    __syncthreads();
    v16h pf  = cat8(*(const v8h*)(pb + m * 32 + half * 8),
                    *(const v8h*)(pb + m * 32 + 16 + half * 8));
    v16h vf0 = cat8(*(const v8h*)(v0row + kb + half * 8),
                    *(const v8h*)(v0row + kb + 16 + half * 8));
    v16h vf1 = cat8(*(const v8h*)(v1row + kb + half * 8),
                    *(const v8h*)(v1row + kb + 16 + half * 8));
    __syncthreads();
    o0 = wmma_f16(pf, vf0, o0);
    o1 = wmma_f16(pf, vf1, o1);
  }
  float* ob = ws + OFF_O;
#pragma unroll
  for (int r = 0; r < 8; ++r) {
    // softmax denominator = PV column 15 of o1 (V dim31 == 1): broadcast it
    float l = __shfl(o1[r], half * 16 + 15);
    float inv = 1.f / l;
    int t = qb * 16 + r + 8 * half;
    int tok = b * TPB + t;
    ob[(size_t)tok * FDIM + hd * 16 + m] = o0[r] * inv;
    if (m < 3) ob[(size_t)tok * FDIM + 128 + hd * 3 + m] = o1[r] * inv;
  }
}

// ---------------- 7) GP / join gating of the MLP --------------------------
__global__ void ga_gate(float* ws) {
  int id = blockIdx.x * blockDim.x + threadIdx.x;
  int tok = id / 9, unit = id % 9;
  if (tok >= BT) return;
  const float* ur = ws + OFF_U + (size_t)tok * M1OUT;
  float* gr = ws + OFF_G + (size_t)tok * FDIM;
  if (unit == 8) {
    for (int s = 0; s < 24; ++s) gr[128 + s] = geluf(ur[256 + s]);
    for (int f = 152; f < 160; ++f) gr[f] = 0.f;
    return;
  }
  float a[16], bb[16], r[16];
  int c = unit & 3;
  int ia = (unit < 4) ? c : 8 + c;
  int ib = (unit < 4) ? 4 + c : 12 + c;
#pragma unroll
  for (int i = 0; i < 16; ++i) { a[i] = ur[ia * 16 + i]; bb[i] = ur[ib * 16 + i]; r[i] = 0.f; }
  if (unit < 4) {                          // geometric product (metric e0^2=0)
    for (int i = 0; i < 16; ++i) {
      int ma = BMASK[i]; float av = a[i];
#pragma unroll
      for (int j = 0; j < 16; ++j) {
        int mb = BMASK[j];
        if (ma & mb & 1) continue;
        float t = av * bb[j];
        r[M2I[ma ^ mb]] += reorder_par(ma, mb) ? -t : t;
      }
    }
  } else {                                 // join = dual(wedge(dual a, dual b))
    float da[16], db[16], w[16];
#pragma unroll
    for (int q = 0; q < 16; ++q) {
      int mq = BMASK[q]; int cm = 15 & ~mq;
      float sg = reorder_par(mq, cm) ? -1.f : 1.f;
      da[M2I[cm]] = sg * a[q]; db[M2I[cm]] = sg * bb[q];
    }
#pragma unroll
    for (int i = 0; i < 16; ++i) w[i] = 0.f;
    for (int i = 0; i < 16; ++i) {
      int mi = BMASK[i]; float av = da[i];
#pragma unroll
      for (int j = 0; j < 16; ++j) {
        int mj = BMASK[j];
        if (mi & mj) continue;
        float t = av * db[j];
        w[M2I[mi | mj]] += reorder_par(mi, mj) ? -t : t;
      }
    }
#pragma unroll
    for (int pc = 0; pc < 16; ++pc) {
      int mp = BMASK[pc];
      float sg = reorder_par(mp, 15 & ~mp) ? -1.f : 1.f;
      r[pc] = sg * w[M2I[15 & ~mp]];
    }
  }
  float gt = geluf(r[0]);
#pragma unroll
  for (int pc = 0; pc < 16; ++pc) gr[unit * 16 + pc] = r[pc] * gt;
}

// ---------------- 8) final projection + output extraction ----------------
__global__ void ga_final(const float* __restrict__ ws, float* __restrict__ out) {
  int tok = blockIdx.x * blockDim.x + threadIdx.x;
  if (tok >= BT) return;
  const float* xr = ws + OFF_H + (size_t)tok * FDIM;
  const float* Wo = ws + OFF_WOUT;
  const float* Bo = ws + OFF_BOUT;
  float acc[16];
#pragma unroll
  for (int p = 0; p < 16; ++p) acc[p] = Bo[p];
  for (int f = 0; f < 152; ++f) {
    float xv = xr[f];
#pragma unroll
    for (int p = 0; p < 16; ++p) acc[p] += xv * Wo[f * 16 + p];
  }
  float d = acc[14];
  if (fabsf(d) <= 0.001f) d = (d >= 0.f) ? 0.001f : -0.001f;
  out[tok * 6 + 0] = -acc[13] / d;
  out[tok * 6 + 1] =  acc[12] / d;
  out[tok * 6 + 2] = -acc[11] / d;
  out[tok * 6 + 3] = 2.f * acc[5];
  out[tok * 6 + 4] = 2.f * acc[6];
  out[tok * 6 + 5] = 2.f * acc[7];
}

extern "C" void kernel_launch(void* const* d_in, const int* in_sizes, int n_in,
                              void* d_out, int out_size, void* d_ws, size_t ws_size,
                              hipStream_t stream) {
  (void)in_sizes; (void)n_in; (void)out_size; (void)ws_size;
  Ptrs P;
  P.x         = (const float*)d_in[0];
  P.w_in_mv   = (const float*)d_in[1];
  P.b_in_mv   = (const float*)d_in[2];
  P.w_in_m2s  = (const float*)d_in[3];
  P.b_in_s    = (const float*)d_in[4];
  P.w_qkv_mv  = (const float*)d_in[5];
  P.w_qkv_s2mv= (const float*)d_in[6];
  P.w_qkv_m2s = (const float*)d_in[7];
  P.w_qkv_s2s = (const float*)d_in[8];
  P.w_ao_mv   = (const float*)d_in[9];
  P.w_ao_s2mv = (const float*)d_in[10];
  P.w_ao_m2s  = (const float*)d_in[11];
  P.w_ao_s2s  = (const float*)d_in[12];
  P.b_ao_mv   = (const float*)d_in[13];
  P.b_ao_s    = (const float*)d_in[14];
  P.w_m1_mv   = (const float*)d_in[15];
  P.w_m1_s2mv = (const float*)d_in[16];
  P.w_m1_m2s  = (const float*)d_in[17];
  P.w_m1_s2s  = (const float*)d_in[18];
  P.b_m1_mv   = (const float*)d_in[19];
  P.b_m1_s    = (const float*)d_in[20];
  P.w_m2_mv   = (const float*)d_in[21];
  P.w_m2_s2mv = (const float*)d_in[22];
  P.w_m2_m2s  = (const float*)d_in[23];
  P.w_m2_s2s  = (const float*)d_in[24];
  P.b_m2_mv   = (const float*)d_in[25];
  P.b_m2_s    = (const float*)d_in[26];
  P.w_out_mv  = (const float*)d_in[27];
  P.w_out_s2mv= (const float*)d_in[28];
  P.b_out_mv  = (const float*)d_in[29];

  float* ws = (float*)d_ws;
  _Float16* hb = (_Float16*)(ws + OFF_END);
  float* out = (float*)d_out;

  dim3 blk256(256), blkG(128);
  dim3 tokGrid(BT / 256);
  dim3 g160(BT / 16, 3);   // 10 N-tiles, 4 waves/block
  dim3 g288(BT / 16, 5);   // 18 N-tiles

  ga_prep  <<<7, 256, 0, stream>>>(P, ws, hb);
  ga_embed <<<tokGrid, blk256, 0, stream>>>(P, ws);
  ga_norm  <<<tokGrid, blk256, 0, stream>>>(ws + OFF_H, ws + OFF_N);
  ga_gemm<160, false, false><<<g160, blkG, 0, stream>>>(ws + OFF_N, hb + WHQo, nullptr, nullptr, ws + OFF_Q);
  ga_gemm<160, false, false><<<g160, blkG, 0, stream>>>(ws + OFF_N, hb + WHKo, nullptr, nullptr, ws + OFF_K);
  ga_gemm<160, false, false><<<g160, blkG, 0, stream>>>(ws + OFF_N, hb + WHVo, nullptr, nullptr, ws + OFF_V);
  ga_repack<<<tokGrid, blk256, 0, stream>>>(ws, hb);
  ga_attn  <<<dim3(TPB / 16, 8, 2), 32, 0, stream>>>(hb, ws);
  ga_gemm<160, true, true><<<g160, blkG, 0, stream>>>(ws + OFF_O, hb + WHAOo, ws + OFF_BAO, ws + OFF_H, ws + OFF_H);
  ga_norm  <<<tokGrid, blk256, 0, stream>>>(ws + OFF_H, ws + OFF_N);
  ga_gemm<M1OUT, true, false><<<g288, blkG, 0, stream>>>(ws + OFF_N, hb + WHM1o, ws + OFF_BM1, nullptr, ws + OFF_U);
  ga_gate  <<<(BT * 9 + 255) / 256, blk256, 0, stream>>>(ws);
  ga_gemm<160, true, true><<<g160, blkG, 0, stream>>>(ws + OFF_G, hb + WHM2o, ws + OFF_BM2, ws + OFF_H, ws + OFF_H);
  ga_final <<<tokGrid, blk256, 0, stream>>>(ws, out);
}
